// DGCNN_63230508531858
// MI455X (gfx1250) — matrix operand order repaired and checked by reading, not compiled, not run on hardware
//
#include <hip/hip_runtime.h>
#include <cstdint>

#define NPTS 2048
#define BATCH 8
#define KNN 20
#define EPS 1e-5f
#define SLOPE 0.2f

typedef __attribute__((ext_vector_type(16))) _Float16 v16h;
typedef __attribute__((ext_vector_type(8)))  float    v8f;

// ---------------------------------------------------------------------------
// WMMA fragment loaders.  LDS tiles are stored as [row][k] halves with row
// stride `ld`.  Packing follows CDNA5 ISA 7.12.2:
//  A (16x32 f16):  lane l holds row M=l&15;  VGPR v: K = (v>=4?16:0) + (l>=16?8:0) + (v&3)*2
//  B (32x16 f16):  lane l holds col N=l&15;  VGPR v: K = (v>=4?16:0) + (l>=16?8:0) + (v&3)*2
// ---------------------------------------------------------------------------
__device__ inline v16h load_a_frag(const _Float16* a, int lane, int kbase, int ld) {
    int row = lane & 15;
    int koff = (lane & 16) ? 8 : 0;
    v16h f;
#pragma unroll
    for (int v = 0; v < 8; ++v) {
        int k = kbase + ((v & 4) ? 16 : 0) + koff + (v & 3) * 2;
        f[2 * v]     = a[row * ld + k];
        f[2 * v + 1] = a[row * ld + k + 1];
    }
    return f;
}

__device__ inline v16h load_b_frag(const _Float16* b, int lane, int kbase, int ld) {
    int col = lane & 15;
    int koff = (lane & 16) ? 8 : 0;
    v16h f;
#pragma unroll
    for (int v = 0; v < 8; ++v) {
        int k = kbase + ((v & 4) ? 16 : 0) + koff + (v & 3) * 2;
        f[2 * v]     = b[col * ld + k];
        f[2 * v + 1] = b[col * ld + k + 1];
    }
    return f;
}

__device__ inline v8f wmma_f16(v16h a, v16h b, v8f c) {
    return __builtin_amdgcn_wmma_f32_16x16x32_f16(false, a, false, b, (short)0, c,
                                                  false, false);
}

// ---------------------------------------------------------------------------
// Squared norms per point: xx[b][n] = sum_c feat[b][n][c]^2   (feat: B,N,C)
// ---------------------------------------------------------------------------
__global__ __launch_bounds__(256) void sqnorm_kernel(const float* __restrict__ feat,
                                                     float* __restrict__ xx,
                                                     int N, int C) {
    int b = blockIdx.y;
    int n = blockIdx.x * 256 + threadIdx.x;
    if (n >= N) return;
    const float* p = feat + ((size_t)b * N + n) * C;
    float s = 0.f;
    for (int c = 0; c < C; ++c) { float v = p[c]; s += v * v; }
    xx[b * N + n] = s;
}

// ---------------------------------------------------------------------------
// Fused kNN: pd[n][m] = 2*dot(x_n,x_m) - xx[n] - xx[m], running top-20 per row.
// One WG handles 32 rows; 4 waves each compute a 16x16 WMMA tile of a 32x32
// pd tile; wave 0 lanes own one row each and keep a sorted top-20 in VGPRs.
// ---------------------------------------------------------------------------
__global__ __launch_bounds__(128) void knn_kernel(const float* __restrict__ feat,
                                                  const float* __restrict__ xx,
                                                  int* __restrict__ idx,
                                                  int N, int C, int CP) {
    __shared__ _Float16 aT[32 * 128];
    __shared__ _Float16 bT[32 * 128];
    __shared__ float pd[32][33];

    int b  = blockIdx.y;
    int r0 = blockIdx.x * 32;
    int lane = threadIdx.x & 31;
    int wave = threadIdx.x >> 5;
    int bN = b * N;
    const float* fb = feat + (size_t)bN * C;

    for (int i = threadIdx.x; i < 32 * CP; i += 128) {
        int r = i / CP, c = i % CP;
        aT[r * CP + c] = (c < C) ? (_Float16)fb[(size_t)(r0 + r) * C + c] : (_Float16)0.f;
    }

    float bd[KNN];
    int   bi[KNN];
#pragma unroll
    for (int k = 0; k < KNN; ++k) { bd[k] = -3.402823e38f; bi[k] = 0; }

    __syncthreads();

    int tr = (wave >> 1) * 16;
    int tc = (wave & 1) * 16;

    for (int m0 = 0; m0 < N; m0 += 32) {
        for (int i = threadIdx.x; i < 32 * CP; i += 128) {
            int r = i / CP, c = i % CP;
            bT[r * CP + c] = (c < C) ? (_Float16)fb[(size_t)(m0 + r) * C + c]
                                     : (_Float16)0.f;
        }
        if (m0 + 32 < N)
            __builtin_prefetch(fb + (size_t)(m0 + 32) * C + lane, 0, 1);
        __syncthreads();

        v8f acc = {};
        for (int kc = 0; kc < CP; kc += 32) {
            v16h af = load_a_frag(aT + tr * CP, lane, kc, CP);
            v16h bf = load_b_frag(bT + tc * CP, lane, kc, CP);
            acc = wmma_f16(af, bf, acc);
        }
        int col = tc + (lane & 15);
        float cxx = xx[bN + m0 + col];
#pragma unroll
        for (int j = 0; j < 8; ++j) {
            int row = tr + j + ((lane & 16) ? 8 : 0);
            pd[row][col] = 2.f * acc[j] - xx[bN + r0 + row] - cxx;
        }
        __syncthreads();

        if (wave == 0) {
            for (int c = 0; c < 32; ++c) {
                float v = pd[lane][c];
                if (v > bd[KNN - 1]) {
                    int pos = 0;
#pragma unroll
                    for (int p = 0; p < KNN - 1; ++p) pos += (bd[p] >= v) ? 1 : 0;
#pragma unroll
                    for (int p = KNN - 1; p >= 1; --p)
                        if (p > pos) { bd[p] = bd[p - 1]; bi[p] = bi[p - 1]; }
#pragma unroll
                    for (int p = 0; p < KNN; ++p)
                        if (p == pos) { bd[p] = v; bi[p] = m0 + c; }
                }
            }
        }
        __syncthreads();
    }

    if (wave == 0) {
        int* op = idx + ((size_t)bN + r0 + lane) * KNN;
#pragma unroll
        for (int k = 0; k < KNN; ++k) op[k] = bi[k];
    }
}

// ---------------------------------------------------------------------------
// GEMM: out[b,n,o] = sum_c A[b,n,c] * W(o,c)
//   wmode 0: W(o,c) = W[o*ldw + c]                 (w_a, or plain weight)
//   wmode 1: W(o,c) = W[o*ldw + C + c] - W[o*ldw + c]   (w_b - w_a)
//   outT  1: write out[b,o,n] instead (channel-major, for the final layer)
// 32x32 output tile per 128-thread WG, K chunked through LDS in 128-wide slabs.
// ---------------------------------------------------------------------------
__global__ __launch_bounds__(128) void gemm_kernel(const float* __restrict__ A,
                                                   const float* __restrict__ W,
                                                   float* __restrict__ out,
                                                   int N, int C, int CP, int O,
                                                   int ldw, int wmode, int outT) {
    __shared__ _Float16 aT[32 * 128];
    __shared__ _Float16 wT[32 * 128];

    int b  = blockIdx.z;
    int n0 = blockIdx.x * 32;
    int o0 = blockIdx.y * 32;
    int lane = threadIdx.x & 31;
    int wave = threadIdx.x >> 5;
    int tr = (wave >> 1) * 16;   // rows (n)
    int tc = (wave & 1) * 16;    // cols (o)

    const float* Ab = A + (size_t)b * N * C;
    v8f acc = {};

    for (int kc0 = 0; kc0 < CP; kc0 += 128) {
        for (int i = threadIdx.x; i < 32 * 128; i += 128) {
            int r = i >> 7, c = i & 127;
            int kg = kc0 + c;
            float av = 0.f, wv = 0.f;
            if (kg < C) {
                av = Ab[(size_t)(n0 + r) * C + kg];
                const float* wr = W + (size_t)(o0 + r) * ldw;
                wv = wmode ? (wr[C + kg] - wr[kg]) : wr[kg];
            }
            aT[i] = (_Float16)av;
            wT[i] = (_Float16)wv;
        }
        __syncthreads();
        int klen = CP - kc0; if (klen > 128) klen = 128;
        for (int kc = 0; kc < klen; kc += 32) {
            v16h af = load_a_frag(aT + tr * 128, lane, kc, 128);
            v16h bf = load_b_frag(wT + tc * 128, lane, kc, 128);
            acc = wmma_f16(af, bf, acc);
        }
        __syncthreads();
    }

    int col = tc + (lane & 15);
#pragma unroll
    for (int j = 0; j < 8; ++j) {
        int row = tr + j + ((lane & 16) ? 8 : 0);
        int n = n0 + row, o = o0 + col;
        if (outT) out[((size_t)b * O + o) * N + n] = acc[j];
        else      out[((size_t)b * N + n) * O + o] = acc[j];
    }
}

// ---------------------------------------------------------------------------
// Edge-conv stats pass: sum / sumsq over (n,k) of h = g[b, idx[n,k], o] + t[b,n,o]
// per (b,o).  Coalesced: consecutive threads read consecutive o of gathered rows.
// ---------------------------------------------------------------------------
__global__ __launch_bounds__(256) void edge_stats_kernel(const float* __restrict__ g,
                                                         const float* __restrict__ t,
                                                         const int* __restrict__ idx,
                                                         float* __restrict__ stats,
                                                         int N, int O) {
    __shared__ float ls[256], lq[256];
    int b  = blockIdx.y;
    int p0 = blockIdx.x * 16;
    int OT = (O < 256) ? O : 256;
    int o  = threadIdx.x % OT;
    int pg = threadIdx.x / OT;
    int ng = 256 / OT;

    float s = 0.f, q = 0.f;
    for (int p = pg; p < 16; p += ng) {
        int n = p0 + p;
        float tv = t[((size_t)b * N + n) * O + o];
        const int* ip = idx + ((size_t)b * N + n) * KNN;
#pragma unroll 4
        for (int k = 0; k < KNN; ++k) {
            float v = g[((size_t)b * N + ip[k]) * O + o] + tv;
            s += v; q += v * v;
        }
    }
    ls[threadIdx.x] = s; lq[threadIdx.x] = q;
    __syncthreads();
    if (pg == 0) {
        for (int gi = 1; gi < ng; ++gi) { s += ls[gi * OT + o]; q += lq[gi * OT + o]; }
        atomicAdd(&stats[(size_t)(b * O + o) * 2 + 0], s);
        atomicAdd(&stats[(size_t)(b * O + o) * 2 + 1], q);
    }
}

// ---------------------------------------------------------------------------
// Edge-conv output pass: xout[b,n,o] = max_k leaky((h - mean) * rsqrt(var+eps))
// ---------------------------------------------------------------------------
__global__ __launch_bounds__(256) void edge_out_kernel(const float* __restrict__ g,
                                                       const float* __restrict__ t,
                                                       const int* __restrict__ idx,
                                                       const float* __restrict__ stats,
                                                       float* __restrict__ xout,
                                                       int N, int O) {
    int b = blockIdx.y;
    size_t e = (size_t)blockIdx.x * 256 + threadIdx.x;
    if (e >= (size_t)N * O) return;
    int o = (int)(e % O);
    int n = (int)(e / O);

    float cnt  = (float)N * (float)KNN;
    float mean = stats[(size_t)(b * O + o) * 2 + 0] / cnt;
    float var  = stats[(size_t)(b * O + o) * 2 + 1] / cnt - mean * mean;
    float rs   = rsqrtf(var + EPS);

    float tv = t[((size_t)b * N + n) * O + o];
    const int* ip = idx + ((size_t)b * N + n) * KNN;
    float m = -3.402823e38f;
#pragma unroll 4
    for (int k = 0; k < KNN; ++k) {
        float v = g[((size_t)b * N + ip[k]) * O + o] + tv;
        float z = (v - mean) * rs;
        z = (z >= 0.f) ? z : SLOPE * z;
        m = fmaxf(m, z);
    }
    xout[((size_t)b * N + n) * O + o] = m;
}

// ---------------------------------------------------------------------------
// Concat x1..x4 channel-wise into cat (B,N,512)
// ---------------------------------------------------------------------------
__global__ __launch_bounds__(256) void concat_kernel(const float* __restrict__ f1,
                                                     const float* __restrict__ f2,
                                                     const float* __restrict__ f3,
                                                     const float* __restrict__ f4,
                                                     float* __restrict__ cat, int N) {
    int b = blockIdx.y;
    size_t e = (size_t)blockIdx.x * 256 + threadIdx.x;
    if (e >= (size_t)N * 512) return;
    int c = (int)(e % 512);
    int n = (int)(e / 512);
    size_t base = (size_t)b * N + n;
    float v;
    if      (c < 64)  v = f1[base * 64  + c];
    else if (c < 128) v = f2[base * 64  + (c - 64)];
    else if (c < 256) v = f3[base * 128 + (c - 128)];
    else              v = f4[base * 256 + (c - 256)];
    cat[base * 512 + c] = v;
}

// ---------------------------------------------------------------------------
// Final instance-norm stats over n (h stored channel-major in d_out: B,512,N)
// ---------------------------------------------------------------------------
__global__ __launch_bounds__(256) void final_stats_kernel(const float* __restrict__ h,
                                                          float* __restrict__ fstats,
                                                          int N) {
    __shared__ float ls[256], lq[256];
    int b = blockIdx.y, o = blockIdx.x;
    const float* p = h + ((size_t)b * 512 + o) * N;
    float s = 0.f, q = 0.f;
    for (int n = threadIdx.x; n < N; n += 256) { float v = p[n]; s += v; q += v * v; }
    ls[threadIdx.x] = s; lq[threadIdx.x] = q;
    __syncthreads();
    for (int off = 128; off > 0; off >>= 1) {
        if (threadIdx.x < off) {
            ls[threadIdx.x] += ls[threadIdx.x + off];
            lq[threadIdx.x] += lq[threadIdx.x + off];
        }
        __syncthreads();
    }
    if (threadIdx.x == 0) {
        fstats[(size_t)(b * 512 + o) * 2 + 0] = ls[0];
        fstats[(size_t)(b * 512 + o) * 2 + 1] = lq[0];
    }
}

// ---------------------------------------------------------------------------
// Final normalize + leaky, in place on d_out
// ---------------------------------------------------------------------------
__global__ __launch_bounds__(256) void final_out_kernel(float* __restrict__ h,
                                                        const float* __restrict__ fstats,
                                                        int N) {
    int b = blockIdx.y;
    size_t e = (size_t)blockIdx.x * 256 + threadIdx.x;
    if (e >= (size_t)512 * N) return;
    int o = (int)(e / N);
    int n = (int)(e % N);
    float mean = fstats[(size_t)(b * 512 + o) * 2 + 0] / (float)N;
    float var  = fstats[(size_t)(b * 512 + o) * 2 + 1] / (float)N - mean * mean;
    float rs   = rsqrtf(var + EPS);
    size_t p = ((size_t)b * 512 + o) * N + n;
    float z = (h[p] - mean) * rs;
    h[p] = (z >= 0.f) ? z : SLOPE * z;
}

__global__ __launch_bounds__(256) void zero_kernel(float* __restrict__ p, int n) {
    int i = blockIdx.x * 256 + threadIdx.x;
    if (i < n) p[i] = 0.f;
}

// ---------------------------------------------------------------------------
extern "C" void kernel_launch(void* const* d_in, const int* in_sizes, int n_in,
                              void* d_out, int out_size, void* d_ws, size_t ws_size,
                              hipStream_t stream) {
    (void)in_sizes; (void)n_in; (void)out_size; (void)ws_size;
    const float* x  = (const float*)d_in[0];
    const float* w1 = (const float*)d_in[1];
    const float* w2 = (const float*)d_in[2];
    const float* w3 = (const float*)d_in[3];
    const float* w4 = (const float*)d_in[4];
    const float* w5 = (const float*)d_in[5];

    const int B = BATCH, N = NPTS;

    // Workspace layout (floats)
    float* ws  = (float*)d_ws;
    float* f1  = ws;                           // B*N*64  = 1048576
    float* f2  = f1  + (size_t)B * N * 64;     // 1048576
    float* f3  = f2  + (size_t)B * N * 64;     // 2097152
    float* f4  = f3  + (size_t)B * N * 128;    // 4194304
    float* gb  = f4  + (size_t)B * N * 256;    // 4194304 (max O=256)
    float* tb  = gb  + (size_t)B * N * 256;    // 4194304
    float* catb = gb;                          // aliases gb..tb (B*N*512) after layer 4
    float* xxb = tb  + (size_t)B * N * 256;    // B*N
    float* st  = xxb + (size_t)B * N;          // B*512*2
    int*  idxb = (int*)(st + (size_t)B * 512 * 2);  // B*N*KNN ints

    struct LayerDesc { const float* feat; int C; int CP; const float* W; int ldw; int O; float* out; };
    LayerDesc L[4] = {
        { x,  3,   32,  w1, 6,   64,  f1 },
        { f1, 64,  64,  w2, 128, 64,  f2 },
        { f2, 64,  64,  w3, 128, 128, f3 },
        { f3, 128, 128, w4, 256, 256, f4 },
    };

    for (int l = 0; l < 4; ++l) {
        const LayerDesc& d = L[l];
        sqnorm_kernel<<<dim3((N + 255) / 256, B), 256, 0, stream>>>(d.feat, xxb, N, d.C);
        knn_kernel<<<dim3(N / 32, B), 128, 0, stream>>>(d.feat, xxb, idxb, N, d.C, d.CP);
        // g = w_a * x    (per-point, point-major output)
        gemm_kernel<<<dim3(N / 32, d.O / 32, B), 128, 0, stream>>>(
            d.feat, d.W, gb, N, d.C, d.CP, d.O, d.ldw, 0, 0);
        // t = (w_b - w_a) * x
        gemm_kernel<<<dim3(N / 32, d.O / 32, B), 128, 0, stream>>>(
            d.feat, d.W, tb, N, d.C, d.CP, d.O, d.ldw, 1, 0);
        zero_kernel<<<(B * d.O * 2 + 255) / 256, 256, 0, stream>>>(st, B * d.O * 2);
        edge_stats_kernel<<<dim3(N / 16, B), 256, 0, stream>>>(gb, tb, idxb, st, N, d.O);
        edge_out_kernel<<<dim3((N * d.O + 255) / 256, B), 256, 0, stream>>>(
            gb, tb, idxb, st, d.out, N, d.O);
    }

    // cat -> h (written channel-major straight into d_out) -> instnorm -> leaky
    concat_kernel<<<dim3((N * 512 + 255) / 256, B), 256, 0, stream>>>(f1, f2, f3, f4, catb, N);
    gemm_kernel<<<dim3(N / 32, 512 / 32, B), 128, 0, stream>>>(
        catb, w5, (float*)d_out, N, 512, 512, 512, 512, 0, 1);
    final_stats_kernel<<<dim3(512, B), 256, 0, stream>>>((const float*)d_out, st, N);
    final_out_kernel<<<dim3((512 * N + 255) / 256, B), 256, 0, stream>>>(
        (float*)d_out, st, N);
}
